// LSTM_11742440587420
// MI455X (gfx1250) — matrix operand order repaired and checked by reading, not compiled, not run on hardware
//
#include <hip/hip_runtime.h>
#include <hip/hip_bf16.h>

// ---------------------------------------------------------------------------
// Persistent wave32/WMMA LSTM for gfx1250 (MI455X).
//   gates_t = [X_t | H_{t-1}] @ [Wx;Wh]  (K=1536, 4096 cols, bf16 WMMA, f32 acc)
//   64 persistent WGs x 256 threads; each WG owns 16 H-columns (64 gate cols),
//   weight slices LDS-resident for all 512 steps, C-state LDS-resident,
//   A-operand staged via GLOBAL_LOAD_ASYNC_TO_LDS_B128 (ASYNCcnt, 3 buffers),
//   one grid-wide sync per timestep. Y_{t-1} GEMM fused into step t's K-loop.
// ---------------------------------------------------------------------------

#define T_STEPS 512
#define NB      64
#define DIN     512
#define DH      1024
#define KTOT    1536        // DIN + DH
#define DOUT    512
#define NWG     64
#define NTHR    256
#define GCOLS   64          // gate columns per WG (4 gates x 16 H cols)
#define HCOLS   16
#define KSTAGE  64          // K per staged LDS chunk (2 WMMA k-steps)
#define NSTAGE  (KTOT / KSTAGE)  // 24 (X: 0..7, H: 8..23)
#define ABUF    (NB * KSTAGE)    // 4096 bf16 per stage buffer

typedef __attribute__((ext_vector_type(16))) __bf16 bf16x16;
typedef __attribute__((ext_vector_type(8)))  __bf16 bf16x8;
typedef __attribute__((ext_vector_type(8)))  float  f32x8;

// ---- workspace layout (bytes) ----
#define OFF_CTR  ((size_t)0)
#define OFF_HB   ((size_t)256)                             // 2*NB*DH bf16
#define OFF_WP   (OFF_HB + (size_t)2*NB*DH*2)              // 64*64*1536 bf16
#define OFF_WQ   (OFF_WP + (size_t)NWG*GCOLS*KTOT*2)       // 512*1024 bf16
#define OFF_XB   (OFF_WQ + (size_t)DOUT*DH*2)              // T*NB*DIN bf16

// ---- dynamic LDS layout (bytes) ----
#define SM_W     0                       // 64*1536 bf16 = 196608
#define SM_WQ    196608                  // 16*1024 bf16 =  32768
#define SM_A     229376                  // 3 * 4096 bf16 = 24576 (triple buffer)
#define SM_G     253952                  // 64*64 f32 = 16384
#define SM_C     270336                  // 64*16 f32 =  4096
#define SMEM_BYTES 274432

__device__ __forceinline__ float sigmoidf_(float x) {
  return 1.0f / (1.0f + __expf(-x));
}

// ---- CDNA5 async global->LDS (ASYNCcnt-tracked); generic shared pointer's
// low 32 bits are the LDS byte offset (aperture convention, ISA 10.2).
__device__ __forceinline__ void async_ld_b128(void* lds_dst, const void* gsrc) {
  unsigned l = (unsigned)(uintptr_t)lds_dst;
  unsigned long long g = (unsigned long long)(uintptr_t)gsrc;
  asm volatile("global_load_async_to_lds_b128 %0, %1, off"
               :: "v"(l), "v"(g) : "memory");
}

// A fragment (16x32 bf16): lane L -> row M=L%16; K = {16h .. 16h+15} contiguous
// Ab points at the 32-wide K slice start; row stride = KSTAGE elements.
__device__ __forceinline__ bf16x16 frag_a(const __bf16* Ab, int rt, int lane) {
  const int h = lane >> 4, m = lane & 15;
  const __bf16* p = Ab + (rt * 16 + m) * KSTAGE + h * 16;
  bf16x8 lo = *(const bf16x8*)(p);
  bf16x8 hi = *(const bf16x8*)(p + 8);
  bf16x16 r;
#pragma unroll
  for (int i = 0; i < 8; ++i) { r[i] = lo[i]; r[i + 8] = hi[i]; }
  return r;
}

// B fragment (32x16 bf16), column-major LDS [col][k]:
// lane L -> col N=L%16; K = {16h .. 16h+15} contiguous.
__device__ __forceinline__ bf16x16 frag_b(const __bf16* Bk, int colstride, int lane) {
  const int h = lane >> 4, c = lane & 15;
  const __bf16* p = Bk + (size_t)c * colstride + h * 16;
  bf16x8 lo = *(const bf16x8*)(p);
  bf16x8 hi = *(const bf16x8*)(p + 8);
  bf16x16 r;
#pragma unroll
  for (int i = 0; i < 8; ++i) { r[i] = lo[i]; r[i + 8] = hi[i]; }
  return r;
}

// Stage one [64 x 64] bf16 A-chunk into LDS: 32B per thread = 2 async b128.
__device__ __forceinline__ void stage_async(const __bf16* __restrict__ Xb,
                                            const __bf16* __restrict__ Hprev,
                                            int t, int st, __bf16* dst, int tid) {
  const int row = tid >> 2, part = tid & 3;
  const __bf16* src = (st < 8)
      ? (Xb + ((size_t)t * NB + row) * DIN + st * KSTAGE + part * 16)
      : (Hprev + (size_t)row * DH + (st - 8) * KSTAGE + part * 16);
  __bf16* d = dst + row * KSTAGE + part * 16;
  async_ld_b128(d, src);
  async_ld_b128(d + 8, src + 8);
}

__global__ void __launch_bounds__(NTHR, 1)
lstm_persistent(const __bf16* __restrict__ Wp, const __bf16* __restrict__ Wqp,
                const __bf16* __restrict__ Xb, __bf16* __restrict__ Hb,
                const float* __restrict__ b_i, const float* __restrict__ b_f,
                const float* __restrict__ b_o, const float* __restrict__ b_c,
                const float* __restrict__ b_q,
                float* __restrict__ out, unsigned* __restrict__ ctr) {
  extern __shared__ char smem[];
  __bf16* Wlds  = (__bf16*)(smem + SM_W);
  __bf16* Wqlds = (__bf16*)(smem + SM_WQ);
  __bf16* Alds  = (__bf16*)(smem + SM_A);
  float*  Glds  = (float*) (smem + SM_G);
  float*  Clds  = (float*) (smem + SM_C);

  const int wg   = blockIdx.x;
  const int tid  = threadIdx.x;
  const int lane = tid & 31;
  const int wv   = tid >> 5;        // 8 waves (wave32)
  const int half = lane >> 4;
  const int ln   = lane & 15;

  // ---- startup: make this WG's weight slices LDS-resident for all 512 steps
  {
    const float4* sw = (const float4*)(Wp + (size_t)wg * GCOLS * KTOT);
    float4* dw = (float4*)Wlds;
    for (int i = tid; i < GCOLS * KTOT * 2 / 16; i += NTHR) dw[i] = sw[i];
    const float4* sq = (const float4*)(Wqp + (size_t)(wg & 31) * 16 * DH);
    float4* dq = (float4*)Wqlds;
    for (int i = tid; i < 16 * DH * 2 / 16; i += NTHR) dq[i] = sq[i];
  }
  for (int e = tid; e < NB * HCOLS; e += NTHR) Clds[e] = 0.0f;  // C0 = 0

  const unsigned nwg = gridDim.x;
  const int rt  = wv & 3;              // gate output row-tile for this wave
  const int ct0 = (wv >> 2) * 2;       // this wave's two gate col-tiles
  const int yrt = (wg >> 5) * 2 + wv;  // Y row-tile (valid for wv<2)
  const int yct = wg & 31;             // Y col-tile

  // hoist biases out of the 512-step loop
  float bi_[4], bf_[4], bo_[4], bc_[4];
#pragma unroll
  for (int j = 0; j < 4; ++j) {
    const int hcol = wg * HCOLS + ((tid + j * NTHR) & 15);
    bi_[j] = b_i[hcol]; bf_[j] = b_f[hcol];
    bo_[j] = b_o[hcol]; bc_[j] = b_c[hcol];
  }
  float bq_ = 0.0f;
  if (wv < 2) bq_ = b_q[yct * 16 + ln];
  __syncthreads();

  // t == T_STEPS is a tail pass that only finishes Y_{T-1}.
  for (int t = 0; t <= T_STEPS; ++t) {
    const bool do_g = (t < T_STEPS);
    const bool do_y = (t > 0);
    const __bf16* Hprev = Hb + (size_t)((t + 1) & 1) * NB * DH;  // H_{t-1}
    __bf16* Hcur = Hb + (size_t)(t & 1) * NB * DH;               // H_t

    f32x8 acc0 = {}, acc1 = {}, yacc = {};

    const int st0 = do_g ? 0 : 8;
    const int nst = NSTAGE - st0;
    // async pipeline prologue: stages st0, st0+1 into buffers 0,1
    stage_async(Xb, Hprev, t, st0,     Alds,        tid);
    stage_async(Xb, Hprev, t, st0 + 1, Alds + ABUF, tid);

    for (int i = 0; i < nst; ++i) {
      // wait until stage i's 2 async loads (this wave) completed; allow the
      // newer stage to remain in flight (in-order completion per ISA 4.1).
      if (i + 1 < nst) asm volatile("s_wait_asynccnt 0x2" ::: "memory");
      else             asm volatile("s_wait_asynccnt 0x0" ::: "memory");
      __syncthreads();  // stage i visible to all waves; frees buffer (i+2)%3
      if (i + 2 < nst)
        stage_async(Xb, Hprev, t, st0 + i + 2, Alds + ((i + 2) % 3) * ABUF, tid);

      const __bf16* Ab = Alds + (i % 3) * ABUF;
      const int st = st0 + i;
#pragma unroll
      for (int sub = 0; sub < 2; ++sub) {
        const int kc = st * 2 + sub;          // 32-wide K index
        const __bf16* Asub = Ab + sub * 32;
        if (do_g) {
          bf16x16 a  = frag_a(Asub, rt, lane);
          bf16x16 b0 = frag_b(Wlds + (size_t)(ct0 * 16) * KTOT + kc * 32, KTOT, lane);
          acc0 = __builtin_amdgcn_wmma_f32_16x16x32_bf16(false, a, false, b0,
                                                         (short)0, acc0, false, false);
          bf16x16 b1 = frag_b(Wlds + (size_t)((ct0 + 1) * 16) * KTOT + kc * 32, KTOT, lane);
          acc1 = __builtin_amdgcn_wmma_f32_16x16x32_bf16(false, a, false, b1,
                                                         (short)0, acc1, false, false);
        }
        if (do_y && wv < 2 && st >= 8) {   // Y_{t-1} = H_{t-1} @ W_hq (fused)
          bf16x16 ay = frag_a(Asub, yrt, lane);
          bf16x16 by = frag_b(Wqlds + (size_t)(kc - 16) * 32, DH, lane);
          yacc = __builtin_amdgcn_wmma_f32_16x16x32_bf16(false, ay, false, by,
                                                         (short)0, yacc, false, false);
        }
      }
    }

    if (do_y && wv < 2) {  // store Y_{t-1} tile (+ bias)
#pragma unroll
      for (int r = 0; r < 8; ++r) {
        const int row = yrt * 16 + r + 8 * half;
        const int col = yct * 16 + ln;
        out[(size_t)(t - 1) * NB * DOUT + (size_t)row * DOUT + col] = yacc[r] + bq_;
      }
    }

    if (do_g) {
      // spill gate accumulators -> LDS (C/D layout: row = r + 8*half, col = ln)
#pragma unroll
      for (int r = 0; r < 8; ++r) {
        Glds[(rt * 16 + r + 8 * half) * GCOLS + (ct0    ) * 16 + ln] = acc0[r];
        Glds[(rt * 16 + r + 8 * half) * GCOLS + (ct0 + 1) * 16 + ln] = acc1[r];
      }
      __syncthreads();

      // element-wise LSTM cell update on this WG's 16 H-columns
#pragma unroll
      for (int j = 0; j < 4; ++j) {
        const int e = tid + j * NTHR;
        const int row = e >> 4, col = e & 15;
        const int hcol = wg * HCOLS + col;
        const float* g = Glds + row * GCOLS;
        const float I  = sigmoidf_(g[col]      + bi_[j]);
        const float F  = sigmoidf_(g[16 + col] + bf_[j]);
        const float O  = sigmoidf_(g[32 + col] + bo_[j]);
        const float Ct = tanhf(    g[48 + col] + bc_[j]);
        const float c  = F * Clds[e] + I * Ct;
        Clds[e] = c;
        const float h = O * tanhf(c);
        Hcur[(size_t)row * DH + hcol] = (__bf16)h;
        if (t == T_STEPS - 1) {  // final (H, C) outputs, fp32
          out[(size_t)T_STEPS * NB * DOUT + (size_t)row * DH + hcol] = h;
          out[(size_t)T_STEPS * NB * DOUT + (size_t)NB * DH + (size_t)row * DH + hcol] = c;
        }
      }

      // ---- one grid-wide sync per step: publish H_t, wait for all WGs ----
      __syncthreads();
      if (tid == 0) {
        __threadfence();                       // release H_t
        atomicAdd(ctr, 1u);
        const unsigned target = (unsigned)(t + 1) * nwg;
        while (__hip_atomic_load(ctr, __ATOMIC_RELAXED, __HIP_MEMORY_SCOPE_AGENT) < target)
          __builtin_amdgcn_s_sleep(1);
      }
      __syncthreads();
      __threadfence();                         // acquire other WGs' H_t
    }
  }
}

// ---------------------------------------------------------------------------
// Packing / init kernels
// ---------------------------------------------------------------------------

// Fused gate weights, pre-tiled per WG: Wp[wg][c=0..63][k=0..1535] bf16,
// c: gate = c/16 (i,f,o,c), hcol = wg*16 + c%16; k<512 -> W_x, else W_h.
__global__ void pack_w_kernel(const float* __restrict__ Wxi, const float* __restrict__ Whi,
                              const float* __restrict__ Wxf, const float* __restrict__ Whf,
                              const float* __restrict__ Wxo, const float* __restrict__ Who,
                              const float* __restrict__ Wxc, const float* __restrict__ Whc,
                              __bf16* __restrict__ Wp) {
  const size_t idx = (size_t)blockIdx.x * blockDim.x + threadIdx.x;
  const size_t total = (size_t)NWG * GCOLS * KTOT;
  if (idx >= total) return;
  const int k  = (int)(idx % KTOT);
  const int c  = (int)((idx / KTOT) % GCOLS);
  const int wg = (int)(idx / ((size_t)KTOT * GCOLS));
  const int gate = c >> 4;
  const int hcol = wg * 16 + (c & 15);
  const float* Wx = (gate == 0) ? Wxi : (gate == 1) ? Wxf : (gate == 2) ? Wxo : Wxc;
  const float* Wh = (gate == 0) ? Whi : (gate == 1) ? Whf : (gate == 2) ? Who : Whc;
  const float v = (k < DIN) ? Wx[(size_t)k * DH + hcol] : Wh[(size_t)(k - DIN) * DH + hcol];
  Wp[idx] = (__bf16)v;
}

// W_hq transposed to column-major bf16: Wqp[col][k]
__global__ void pack_wq_kernel(const float* __restrict__ Whq, __bf16* __restrict__ Wqp) {
  const size_t idx = (size_t)blockIdx.x * blockDim.x + threadIdx.x;
  if (idx >= (size_t)DOUT * DH) return;
  const int col = (int)(idx / DH);
  const int k   = (int)(idx % DH);
  Wqp[idx] = (__bf16)Whq[(size_t)k * DOUT + col];
}

__global__ void pack_x_kernel(const float* __restrict__ X, __bf16* __restrict__ Xb) {
  const size_t idx = (size_t)blockIdx.x * blockDim.x + threadIdx.x;
  if (idx >= (size_t)T_STEPS * NB * DIN) return;
  Xb[idx] = (__bf16)X[idx];
}

__global__ void init_kernel(__bf16* __restrict__ Hb, unsigned* __restrict__ ctr) {
  const int idx = blockIdx.x * blockDim.x + threadIdx.x;
  if (idx < 2 * NB * DH) Hb[idx] = (__bf16)0.0f;  // H_0 = 0 (both buffers)
  if (idx == 0) *ctr = 0u;                        // sync counter reset every call
}

// ---------------------------------------------------------------------------
extern "C" void kernel_launch(void* const* d_in, const int* in_sizes, int n_in,
                              void* d_out, int out_size, void* d_ws, size_t ws_size,
                              hipStream_t stream) {
  (void)in_sizes; (void)n_in; (void)out_size; (void)ws_size;
  const float* inputs = (const float*)d_in[0];
  const float* W_xi = (const float*)d_in[1];
  const float* W_hi = (const float*)d_in[2];
  const float* b_i  = (const float*)d_in[3];
  const float* W_xf = (const float*)d_in[4];
  const float* W_hf = (const float*)d_in[5];
  const float* b_f  = (const float*)d_in[6];
  const float* W_xo = (const float*)d_in[7];
  const float* W_ho = (const float*)d_in[8];
  const float* b_o  = (const float*)d_in[9];
  const float* W_xc = (const float*)d_in[10];
  const float* W_hc = (const float*)d_in[11];
  const float* b_c  = (const float*)d_in[12];
  const float* W_hq = (const float*)d_in[13];
  const float* b_q  = (const float*)d_in[14];

  char* ws = (char*)d_ws;   // ~46 MB used
  unsigned* ctr = (unsigned*)(ws + OFF_CTR);
  __bf16* Hb  = (__bf16*)(ws + OFF_HB);
  __bf16* Wp  = (__bf16*)(ws + OFF_WP);
  __bf16* Wqp = (__bf16*)(ws + OFF_WQ);
  __bf16* Xb  = (__bf16*)(ws + OFF_XB);

  init_kernel<<<(2 * NB * DH + 255) / 256, 256, 0, stream>>>(Hb, ctr);
  pack_w_kernel<<<(int)(((size_t)NWG * GCOLS * KTOT + 255) / 256), 256, 0, stream>>>(
      W_xi, W_hi, W_xf, W_hf, W_xo, W_ho, W_xc, W_hc, Wp);
  pack_wq_kernel<<<(DOUT * DH + 255) / 256, 256, 0, stream>>>(W_hq, Wqp);
  pack_x_kernel<<<(int)(((size_t)T_STEPS * NB * DIN + 255) / 256), 256, 0, stream>>>(inputs, Xb);

  (void)hipFuncSetAttribute((const void*)lstm_persistent,
                            hipFuncAttributeMaxDynamicSharedMemorySize, SMEM_BYTES);
  lstm_persistent<<<NWG, NTHR, SMEM_BYTES, stream>>>(
      Wp, Wqp, Xb, Hb, b_i, b_f, b_o, b_c, b_q, (float*)d_out, ctr);
}
